// GATModel_4587025072859
// MI455X (gfx1250) — matrix operand order repaired and checked by reading, not compiled, not run on hardware
//
#include <hip/hip_runtime.h>
#include <hip/hip_bf16.h>

typedef __attribute__((ext_vector_type(16))) _Float16 v16h;
typedef __attribute__((ext_vector_type(8)))  _Float16 v8h;
typedef __attribute__((ext_vector_type(8)))  float    v8f;

#define GAT_NEG_SLOPE 0.2f

// ---------------------------------------------------------------- utilities

__global__ void fill_f32_kernel(float* __restrict__ p, float v, long n) {
    long i = (long)blockIdx.x * blockDim.x + threadIdx.x;
    long stride = (long)gridDim.x * blockDim.x;
    for (; i < n; i += stride) p[i] = v;
}

// Convert src[0..nsrc) to f16; zero-fill [nsrc, ntot) (pad rows for the 32-row GEMM tile).
__global__ void cvt_f16_pad_kernel(_Float16* __restrict__ dst, const float* __restrict__ src,
                                   long nsrc, long ntot) {
    long i = (long)blockIdx.x * blockDim.x + threadIdx.x;
    long stride = (long)gridDim.x * blockDim.x;
    for (; i < ntot; i += stride) dst[i] = (i < nsrc) ? (_Float16)src[i] : (_Float16)0.0f;
}

// dst[c*rows + r] = (f16) src[r*cols + c]   (transpose + convert)
__global__ void transpose_cvt_kernel(_Float16* __restrict__ dst, const float* __restrict__ src,
                                     int rows, int cols) {
    long i = (long)blockIdx.x * blockDim.x + threadIdx.x;
    long n = (long)rows * cols;
    if (i >= n) return;
    int r = (int)(i / cols);
    int c = (int)(i % cols);
    dst[(long)c * rows + r] = (_Float16)src[i];
}

__device__ __forceinline__ float leaky(float v) {
    return v > 0.0f ? v : GAT_NEG_SLOPE * v;
}

__device__ __forceinline__ float atomicMaxF(float* addr, float val) {
    if (val >= 0.0f)
        return __int_as_float(atomicMax((int*)addr, __float_as_int(val)));
    return __uint_as_float(atomicMin((unsigned int*)addr, __float_as_uint(val)));
}

// ---------------------------------------------------------------- WMMA helpers

// A operand (16x32 f16): lanes 0-15 hold K = k..k+15; lanes 16-31 hold K = k+16..k+31.
__device__ __forceinline__ v16h load_a_frag(const _Float16* __restrict__ row, int k, int sel) {
    v8h lo = *(const v8h*)(row + k + sel * 16);
    v8h hi = *(const v8h*)(row + k + sel * 16 + 8);
    v16h a;
#pragma unroll
    for (int i = 0; i < 8; ++i) { a[i] = lo[i]; a[i + 8] = hi[i]; }
    return a;
}

// B operand (32x16 f16): per lane 16 contiguous K at k + sel*16 of the transposed weight row.
__device__ __forceinline__ v16h load_b_frag(const _Float16* __restrict__ rowsel, int k) {
    return *(const v16h*)(rowsel + k);
}

#define WMMA_F16(a, b, c) \
    __builtin_amdgcn_wmma_f32_16x16x32_f16(false, (a), false, (b), (short)0, (c), false, false)

// ---------------------------------------------------------------- WMMA GEMMs

// C[nrows,256] f32 = A[Npad,512] f16 * Bt^T ; Bt[n][k] k-contiguous (pre-transposed W1).
// One block = 4 waves; block owns a 32-row stripe, wave w owns columns [w*64, w*64+64).
// Per wave: 2 A-tiles x 4 B-tiles = 8 accumulators; loads double-buffered one K-step ahead.
__global__ __launch_bounds__(128) void gemm1_wmma_kernel(const _Float16* __restrict__ A,
                                                         const _Float16* __restrict__ Bt,
                                                         float* __restrict__ C, int nrows) {
    const int wave = threadIdx.x >> 5;
    const int lane = threadIdx.x & 31;
    const int r    = lane & 15;
    const int sel  = lane >> 4;
    const int m0   = blockIdx.x * 32;
    const int n0   = wave * 64;

    const _Float16* a0row = A + (long)(m0 + r) * 512;
    const _Float16* a1row = A + (long)(m0 + 16 + r) * 512;
    const _Float16* b0r   = Bt + (long)(n0 +  0 + r) * 512 + sel * 16;
    const _Float16* b1r   = Bt + (long)(n0 + 16 + r) * 512 + sel * 16;
    const _Float16* b2r   = Bt + (long)(n0 + 32 + r) * 512 + sel * 16;
    const _Float16* b3r   = Bt + (long)(n0 + 48 + r) * 512 + sel * 16;

    v8f c00 = {}, c01 = {}, c02 = {}, c03 = {};
    v8f c10 = {}, c11 = {}, c12 = {}, c13 = {};

    // prologue: iteration 0 operands
    v16h a0 = load_a_frag(a0row, 0, sel);
    v16h a1 = load_a_frag(a1row, 0, sel);
    v16h b0 = load_b_frag(b0r, 0);
    v16h b1 = load_b_frag(b1r, 0);
    v16h b2 = load_b_frag(b2r, 0);
    v16h b3 = load_b_frag(b3r, 0);

#pragma unroll
    for (int ki = 0; ki < 16; ++ki) {
        v16h a0n = {}, a1n = {}, b0n = {}, b1n = {}, b2n = {}, b3n = {};
        if (ki < 15) {
            const int kn = (ki + 1) * 32;
            a0n = load_a_frag(a0row, kn, sel);
            a1n = load_a_frag(a1row, kn, sel);
            b0n = load_b_frag(b0r, kn);
            b1n = load_b_frag(b1r, kn);
            b2n = load_b_frag(b2r, kn);
            b3n = load_b_frag(b3r, kn);
        }
        c00 = WMMA_F16(a0, b0, c00);
        c01 = WMMA_F16(a0, b1, c01);
        c02 = WMMA_F16(a0, b2, c02);
        c03 = WMMA_F16(a0, b3, c03);
        c10 = WMMA_F16(a1, b0, c10);
        c11 = WMMA_F16(a1, b1, c11);
        c12 = WMMA_F16(a1, b2, c12);
        c13 = WMMA_F16(a1, b3, c13);
        a0 = a0n; a1 = a1n; b0 = b0n; b1 = b1n; b2 = b2n; b3 = b3n;
    }

#pragma unroll
    for (int i = 0; i < 8; ++i) {
        const int row0 = m0 + sel * 8 + i;        // A-tile 0 rows
        const int row1 = row0 + 16;               // A-tile 1 rows
        if (row0 < nrows) {
            long ro = (long)row0 * 256;
            C[ro + n0 +  0 + r] = c00[i];
            C[ro + n0 + 16 + r] = c01[i];
            C[ro + n0 + 32 + r] = c02[i];
            C[ro + n0 + 48 + r] = c03[i];
        }
        if (row1 < nrows) {
            long ro = (long)row1 * 256;
            C[ro + n0 +  0 + r] = c10[i];
            C[ro + n0 + 16 + r] = c11[i];
            C[ro + n0 + 32 + r] = c12[i];
            C[ro + n0 + 48 + r] = c13[i];
        }
    }
}

// C[N,16] f32 = A[N,256] f16 * Bt[16,256] f16 ; one wave per 16-row tile.
__global__ __launch_bounds__(128) void gemm2_wmma_kernel(const _Float16* __restrict__ A,
                                                         const _Float16* __restrict__ Bt,
                                                         float* __restrict__ C, int nrows) {
    const int wave = threadIdx.x >> 5;
    const int lane = threadIdx.x & 31;
    const int mt   = blockIdx.x * 4 + wave;
    const int m0   = mt * 16;
    if (m0 >= nrows) return;                 // wave-uniform exit: EXEC stays all-ones
    const int r   = lane & 15;
    const int sel = lane >> 4;

    const _Float16* arow = A + (long)(m0 + r) * 256;
    const _Float16* brow = Bt + (long)r * 256 + sel * 16;

    v8f c = {};
    v16h a = load_a_frag(arow, 0, sel);
    v16h b = load_b_frag(brow, 0);
#pragma unroll
    for (int ki = 0; ki < 8; ++ki) {
        v16h an = {}, bn = {};
        if (ki < 7) {
            const int kn = (ki + 1) * 32;
            an = load_a_frag(arow, kn, sel);
            bn = load_b_frag(brow, kn);
        }
        c = WMMA_F16(a, b, c);
        a = an; b = bn;
    }
#pragma unroll
    for (int i = 0; i < 8; ++i)
        C[(long)(m0 + sel * 8 + i) * 16 + r] = c[i];
}

// ---------------------------------------------------------------- GAT pieces

// es[n,h] = sum_f feat[n, h*F+f] * a_src[h,f] ; ed likewise.
__global__ void scores_kernel(const float* __restrict__ feat, const float* __restrict__ a_src,
                              const float* __restrict__ a_dst, float* __restrict__ es,
                              float* __restrict__ ed, int n, int H, int F) {
    int idx = blockIdx.x * blockDim.x + threadIdx.x;
    if (idx >= n * H) return;
    int node = idx / H, h = idx % H;
    const float* base = feat + (long)node * H * F + (long)h * F;
    float s = 0.0f, d = 0.0f;
    for (int f = 0; f < F; ++f) {
        float v = base[f];
        s += v * a_src[h * F + f];
        d += v * a_dst[h * F + f];
    }
    es[idx] = s;
    ed[idx] = d;
}

__global__ void edge_max_kernel(const int* __restrict__ src, const int* __restrict__ dst,
                                const float* __restrict__ es, const float* __restrict__ ed,
                                float* __restrict__ m, int E, int H) {
    int tid = blockIdx.x * blockDim.x + threadIdx.x;
    if (tid >= E * H) return;
    int e = tid / H, h = tid % H;
    int s = src[e], d = dst[e];
    float v = leaky(es[s * H + h] + ed[d * H + h]);
    atomicMaxF(&m[d * H + h], v);
}

__global__ void edge_sum_kernel(const int* __restrict__ src, const int* __restrict__ dst,
                                const float* __restrict__ es, const float* __restrict__ ed,
                                const float* __restrict__ m, float* __restrict__ ssum,
                                int E, int H) {
    int tid = blockIdx.x * blockDim.x + threadIdx.x;
    if (tid >= E * H) return;
    int e = tid / H, h = tid % H;
    int s = src[e], d = dst[e];
    float v = leaky(es[s * H + h] + ed[d * H + h]);
    float p = __expf(v - m[d * H + h]);
    atomicAdd(&ssum[d * H + h], p);
}

__global__ void edge_agg_kernel(const int* __restrict__ src, const int* __restrict__ dst,
                                const float* __restrict__ es, const float* __restrict__ ed,
                                const float* __restrict__ m, const float* __restrict__ ssum,
                                const float* __restrict__ feat, float* __restrict__ agg,
                                int E, int H, int F) {
    int tid = blockIdx.x * blockDim.x + threadIdx.x;
    if (tid >= E * H) return;
    int e = tid / H, h = tid % H;
    int s = src[e], d = dst[e];
    float v = leaky(es[s * H + h] + ed[d * H + h]);
    float p = __expf(v - m[d * H + h]);
    float den = ssum[d * H + h];
    float alpha = p / (den > 0.0f ? den : 1.0f);
    const float* fb = feat + (long)s * H * F + (long)h * F;
    float* ob = agg + (long)d * H * F + (long)h * F;
    for (int f = 0; f < F; ++f)
        atomicAdd(&ob[f], fb[f] * alpha);
}

__global__ void elu_cvt_kernel(const float* __restrict__ src, _Float16* __restrict__ dst, long n) {
    long i = (long)blockIdx.x * blockDim.x + threadIdx.x;
    long stride = (long)gridDim.x * blockDim.x;
    for (; i < n; i += stride) {
        float v = src[i];
        float e = (v > 0.0f) ? v : (__expf(v) - 1.0f);
        dst[i] = (_Float16)e;
    }
}

__global__ void logsoftmax_kernel(const float* __restrict__ agg, float* __restrict__ out, int n) {
    int node = blockIdx.x * blockDim.x + threadIdx.x;
    if (node >= n) return;
    const float* row = agg + (long)node * 16;
    float mx = -1e30f;
#pragma unroll
    for (int c = 0; c < 16; ++c) mx = fmaxf(mx, row[c]);
    float sum = 0.0f;
#pragma unroll
    for (int c = 0; c < 16; ++c) sum += __expf(row[c] - mx);
    float lse = mx + __logf(sum);
#pragma unroll
    for (int c = 0; c < 16; ++c) out[(long)node * 16 + c] = row[c] - lse;
}

// ---------------------------------------------------------------- launch

extern "C" void kernel_launch(void* const* d_in, const int* in_sizes, int n_in,
                              void* d_out, int out_size, void* d_ws, size_t ws_size,
                              hipStream_t stream) {
    const float* x    = (const float*)d_in[0];   // [N, 512]
    const float* W1   = (const float*)d_in[1];   // [512, 256]
    const float* a1s  = (const float*)d_in[2];   // [8, 32]
    const float* a1d  = (const float*)d_in[3];   // [8, 32]
    const float* W2   = (const float*)d_in[4];   // [256, 16]
    const float* a2s  = (const float*)d_in[5];   // [1, 16]
    const float* a2d  = (const float*)d_in[6];   // [1, 16]
    const int*   eidx = (const int*)d_in[7];     // [2, E]

    const int N    = in_sizes[0] / 512;          // 50000
    const int Npad = (N + 31) & ~31;             // 50016: pad to 32-row GEMM tiles
    const int E    = in_sizes[7] / 2;            // 800000
    const int* esrc = eidx;
    const int* edst = eidx + E;
    float* out = (float*)d_out;

    // ---- workspace layout (agg1 aliases xh: N*256*4 <= Npad*512*2) ----
    char* ws = (char*)d_ws;
    size_t o = 0;
    _Float16* xh  = (_Float16*)(ws + o);  o += (size_t)Npad * 512 * 2; // f16 x, zero-padded rows
    float*    agg1 = (float*)xh;                                       // alias, reused after GEMM1
    _Float16* w1t = (_Float16*)(ws + o);  o += (size_t)256 * 512 * 2;  // W1^T f16
    float*    h1  = (float*)(ws + o);     o += (size_t)N * 256 * 4;    // layer-1 GEMM out
    float*    es1 = (float*)(ws + o);     o += (size_t)N * 8 * 4;
    float*    ed1 = (float*)(ws + o);     o += (size_t)N * 8 * 4;
    float*    m1  = (float*)(ws + o);     o += (size_t)N * 8 * 4;
    float*    s1  = (float*)(ws + o);     o += (size_t)N * 8 * 4;
    _Float16* z   = (_Float16*)(ws + o);  o += (size_t)N * 256 * 2;    // elu(agg1) f16
    _Float16* w2t = (_Float16*)(ws + o);  o += (size_t)16 * 256 * 2;   // W2^T f16
    float*    h2  = (float*)(ws + o);     o += (size_t)N * 16 * 4;
    float*    es2 = (float*)(ws + o);     o += (size_t)N * 4;
    float*    ed2 = (float*)(ws + o);     o += (size_t)N * 4;
    float*    m2  = (float*)(ws + o);     o += (size_t)N * 4;
    float*    s2  = (float*)(ws + o);     o += (size_t)N * 4;
    float*    agg2 = (float*)(ws + o);    o += (size_t)N * 16 * 4;
    (void)ws_size; (void)n_in; (void)out_size;

    const int TB = 256;
    auto blocks = [](long n, int tb) { return (int)((n + tb - 1) / tb); };

    // 1) conversions / transposes
    cvt_f16_pad_kernel<<<4096, TB, 0, stream>>>(xh, x, (long)N * 512, (long)Npad * 512);
    transpose_cvt_kernel<<<blocks((long)512 * 256, TB), TB, 0, stream>>>(w1t, W1, 512, 256);
    transpose_cvt_kernel<<<blocks((long)256 * 16, TB), TB, 0, stream>>>(w2t, W2, 256, 16);

    // 2) layer-1 GEMM (WMMA f16 -> f32), 32x256 stripe per block, stores guarded to N rows
    gemm1_wmma_kernel<<<Npad / 32, 128, 0, stream>>>(xh, w1t, h1, N);

    // 3) per-node attention scores
    scores_kernel<<<blocks((long)N * 8, TB), TB, 0, stream>>>(h1, a1s, a1d, es1, ed1, N, 8, 32);

    // 4) init softmax/aggregate state (xh dead after GEMM1 -> agg1 reuse is safe)
    fill_f32_kernel<<<blocks((long)N * 8, TB), TB, 0, stream>>>(m1, -1e30f, (long)N * 8);
    fill_f32_kernel<<<blocks((long)N * 8, TB), TB, 0, stream>>>(s1, 0.0f, (long)N * 8);
    fill_f32_kernel<<<4096, TB, 0, stream>>>(agg1, 0.0f, (long)N * 256);

    // 5) layer-1 edge softmax + aggregation (L2-resident atomics: h1 = 51 MB < 192 MB L2)
    edge_max_kernel<<<blocks((long)E * 8, TB), TB, 0, stream>>>(esrc, edst, es1, ed1, m1, E, 8);
    edge_sum_kernel<<<blocks((long)E * 8, TB), TB, 0, stream>>>(esrc, edst, es1, ed1, m1, s1, E, 8);
    edge_agg_kernel<<<blocks((long)E * 8, TB), TB, 0, stream>>>(esrc, edst, es1, ed1, m1, s1,
                                                                h1, agg1, E, 8, 32);

    // 6) ELU -> f16, layer-2 GEMM (WMMA)
    elu_cvt_kernel<<<4096, TB, 0, stream>>>(agg1, z, (long)N * 256);
    gemm2_wmma_kernel<<<(N / 16 + 3) / 4, 128, 0, stream>>>(z, w2t, h2, N);

    // 7) layer-2 attention (single head, C=16)
    scores_kernel<<<blocks((long)N, TB), TB, 0, stream>>>(h2, a2s, a2d, es2, ed2, N, 1, 16);
    fill_f32_kernel<<<blocks((long)N, TB), TB, 0, stream>>>(m2, -1e30f, (long)N);
    fill_f32_kernel<<<blocks((long)N, TB), TB, 0, stream>>>(s2, 0.0f, (long)N);
    fill_f32_kernel<<<blocks((long)N * 16, TB), TB, 0, stream>>>(agg2, 0.0f, (long)N * 16);
    edge_max_kernel<<<blocks((long)E, TB), TB, 0, stream>>>(esrc, edst, es2, ed2, m2, E, 1);
    edge_sum_kernel<<<blocks((long)E, TB), TB, 0, stream>>>(esrc, edst, es2, ed2, m2, s2, E, 1);
    edge_agg_kernel<<<blocks((long)E, TB), TB, 0, stream>>>(esrc, edst, es2, ed2, m2, s2,
                                                            h2, agg2, E, 1, 16);

    // 8) log-softmax over 16 classes
    logsoftmax_kernel<<<blocks((long)N, TB), TB, 0, stream>>>(agg2, out, N);
}
